// GAT_81209241633571
// MI455X (gfx1250) — matrix operand order repaired and checked by reading, not compiled, not run on hardware
//
#include <hip/hip_runtime.h>
#include <cstdint>

#define NNODES 50000
#define NEDGES 800000
#define ETOT   (NEDGES + NNODES)   /* 850000 edges incl. self loops */
#define FIN    128
#define NHID   128
#define NHEADS 4
#define NCLASS 64
#define KCAT   (NHID * NHEADS)     /* 512 */
#define SLOPE  0.2f

typedef __attribute__((ext_vector_type(16))) __bf16         v16bf;
typedef __attribute__((ext_vector_type(8)))  float          v8f;
typedef __attribute__((ext_vector_type(8)))  unsigned short v8u16;

// ---------------- helpers ----------------
__device__ __forceinline__ unsigned short f2bf(float f) {
  unsigned int u = __float_as_uint(f);
  u += 0x7FFFu + ((u >> 16) & 1u);      // round-to-nearest-even
  return (unsigned short)(u >> 16);
}
__device__ __forceinline__ float lrelu(float x, float sl) {
  return x > 0.f ? x : sl * x;
}
// monotonic float -> uint mapping so unsigned atomicMax == float max
__device__ __forceinline__ unsigned int fmapmax(float f) {
  unsigned int u = __float_as_uint(f);
  return (u & 0x80000000u) ? ~u : (u | 0x80000000u);
}
__device__ __forceinline__ float funmapmax(unsigned int u) {
  return __uint_as_float((u & 0x80000000u) ? (u & 0x7FFFFFFFu) : ~u);
}
__device__ __forceinline__ void edge_pair(const int* __restrict__ edge, int et,
                                          int& s, int& d) {
  if (et < NEDGES) { s = edge[et]; d = edge[NEDGES + et]; }
  else             { s = et - NEDGES; d = s; }           // self loop
}

// ---------------- utility kernels ----------------
__global__ void gat_zero_u32(unsigned int* __restrict__ p, long n) {
  long i = (long)blockIdx.x * blockDim.x + threadIdx.x;
  if (i < n) p[i] = 0u;
}

__global__ void gat_f32_to_bf16(const float* __restrict__ src,
                                unsigned short* __restrict__ dst, long n) {
  long i = (long)blockIdx.x * blockDim.x + threadIdx.x;
  if (i < n) dst[i] = f2bf(src[i]);
}

// Pack B [batch][K][Nc] (f32 row-major) into WMMA bf16 B-lane layout:
// Bp[(((b*NT+nt)*KS+ks)*32+L)*16+e] = B[b][ks*32+(e/8)*16+(L/16)*8+e%8][nt*16+(L%16)]
__global__ void gat_pack_B(const float* __restrict__ B,
                           unsigned short* __restrict__ Bp,
                           int K, int Nc, int nbatch) {
  const int NT = Nc >> 4, KS = K >> 5;
  long total = (long)nbatch * NT * KS * 512;
  long t = (long)blockIdx.x * blockDim.x + threadIdx.x;
  if (t >= total) return;
  int  e  = (int)(t & 15);
  int  L  = (int)((t >> 4) & 31);
  long r  = t >> 9;
  int  ks = (int)(r % KS); r /= KS;
  int  nt = (int)(r % NT);
  int  b  = (int)(r / NT);
  int  k  = ks * 32 + (e >> 3) * 16 + (L >> 4) * 8 + (e & 7);
  int  n  = nt * 16 + (L & 15);
  Bp[t] = f2bf(B[((long)b * K + k) * Nc + n]);
}

// ---- WMMA GEMM, N-register-blocked, B panel staged in LDS via async copy --
// C[b] = A[b](MxK, bf16 row-major) @ B[b](KxNc, packed). One wave computes a
// 16-row x Nc strip: A loaded once per K-step, NT independent WMMAs.
template <int NT, int KS>
__global__ void __launch_bounds__(256)
gat_wmma_gemm(const unsigned short* __restrict__ A,
              const unsigned short* __restrict__ Bp,
              float* __restrict__ C, int M) {
  constexpr int K  = KS * 32;
  constexpr int Nc = NT * 16;
  __shared__ unsigned short shB[NT * KS * 512];     // bf16 B panel for batch b

  const int b = blockIdx.y;

  // ---- cooperative async copy of the packed B panel into LDS ----
  {
    const unsigned short* Bpan = Bp + (long)b * NT * KS * 512;
    unsigned lds0 = (unsigned)(size_t)(&shB[0]);    // addr[31:0] = LDS offset
    constexpr int CHUNKS = NT * KS * 64;            // 16B chunks
    for (int c = threadIdx.x; c < CHUNKS; c += 256) {
      unsigned   ldsa = lds0 + (unsigned)c * 16u;
      const void* g   = (const char*)Bpan + (size_t)c * 16u;
      asm volatile("global_load_async_to_lds_b128 %0, %1, off"
                   :: "v"(ldsa), "v"(g) : "memory");
    }
    asm volatile("s_wait_asynccnt 0x0" ::: "memory");
  }
  __syncthreads();

  const int lane = threadIdx.x & 31;
  const int widx = threadIdx.x >> 5;
  const int mt   = blockIdx.x * 8 + widx;           // 16-row strip per wave
  const int MT   = M >> 4;
  if (mt < MT) {
    const int hi = lane >> 4, nl = lane & 15;
    const unsigned short* Ab =
        A + (long)b * M * K + ((long)mt * 16 + nl) * K + hi * 8;

    v8f acc[NT];
#pragma unroll
    for (int nt = 0; nt < NT; ++nt) acc[nt] = (v8f){};

    for (int ks = 0; ks < KS; ++ks) {
      v8u16 alo = *(const v8u16*)(Ab + ks * 32);      // K = kk + hi*8 + 0..7
      v8u16 ahi = *(const v8u16*)(Ab + ks * 32 + 16); // K = kk+16+hi*8 + 0..7
      v16bf av;
#pragma unroll
      for (int i = 0; i < 8; ++i) {
        av[i]     = __builtin_bit_cast(__bf16, (unsigned short)alo[i]);
        av[i + 8] = __builtin_bit_cast(__bf16, (unsigned short)ahi[i]);
      }
#pragma unroll
      for (int nt = 0; nt < NT; ++nt) {
        const unsigned short* Bl = &shB[((nt * KS + ks) * 32 + lane) * 16];
        v8u16 b0 = *(const v8u16*)(Bl);
        v8u16 b1 = *(const v8u16*)(Bl + 8);
        v16bf bv;
#pragma unroll
        for (int i = 0; i < 8; ++i) {
          bv[i]     = __builtin_bit_cast(__bf16, (unsigned short)b0[i]);
          bv[i + 8] = __builtin_bit_cast(__bf16, (unsigned short)b1[i]);
        }
        acc[nt] = __builtin_amdgcn_wmma_f32_16x16x32_bf16(
            /*neg_a=*/false, av, /*neg_b=*/false, bv,
            /*c_mod=*/(short)0, acc[nt], /*reuse_a=*/false, /*reuse_b=*/false);
      }
    }

    float* Crow = C + (long)b * M * Nc + ((long)mt * 16 + hi * 8) * Nc + nl;
#pragma unroll
    for (int nt = 0; nt < NT; ++nt) {
#pragma unroll
      for (int v = 0; v < 8; ++v) Crow[(long)v * Nc + nt * 16] = acc[nt][v];
    }
  }
}

// ---------------- per-node attention scalars: xs . a_src / a_dst ----------
__global__ void gat_dots(const float* __restrict__ xs,
                         const float* __restrict__ asrc,
                         const float* __restrict__ adst,
                         float* __restrict__ ssrc, float* __restrict__ sdst,
                         int F, int nh) {
  long wid  = ((long)blockIdx.x * blockDim.x + threadIdx.x) >> 5;
  int  lane = threadIdx.x & 31;
  if (wid >= (long)nh * NNODES) return;
  int  h    = (int)(wid / NNODES);
  long base = wid * F;
  float ss = 0.f, sd = 0.f;
  for (int f = lane; f < F; f += 32) {
    float x = xs[base + f];
    ss += x * asrc[(long)h * F + f];
    sd += x * adst[(long)h * F + f];
  }
  for (int m = 16; m; m >>= 1) {
    ss += __shfl_xor(ss, m, 32);
    sd += __shfl_xor(sd, m, 32);
  }
  if (lane == 0) { ssrc[wid] = ss; sdst[wid] = sd; }
}

// ---------------- segment max over destinations ---------------------------
__global__ void gat_edge_max(const int* __restrict__ edge,
                             const float* __restrict__ ssrc,
                             const float* __restrict__ sdst,
                             unsigned int* __restrict__ mmax, int nh) {
  long t = (long)blockIdx.x * blockDim.x + threadIdx.x;
  if (t >= (long)nh * ETOT) return;
  int et = (int)(t % ETOT), h = (int)(t / ETOT);
  int s, d; edge_pair(edge, et, s, d);
  float e = lrelu(ssrc[(long)h * NNODES + s] + sdst[(long)h * NNODES + d], SLOPE);
  atomicMax(&mmax[(long)h * NNODES + d], fmapmax(e));
}

// ---------------- exp(e - m) and segment sum ------------------------------
__global__ void gat_edge_ex(const int* __restrict__ edge,
                            const float* __restrict__ ssrc,
                            const float* __restrict__ sdst,
                            const unsigned int* __restrict__ mmax,
                            float* __restrict__ exbuf,
                            float* __restrict__ denom, int nh) {
  long t = (long)blockIdx.x * blockDim.x + threadIdx.x;
  if (t >= (long)nh * ETOT) return;
  int et = (int)(t % ETOT), h = (int)(t / ETOT);
  int s, d; edge_pair(edge, et, s, d);
  float e  = lrelu(ssrc[(long)h * NNODES + s] + sdst[(long)h * NNODES + d], SLOPE);
  float mv = funmapmax(mmax[(long)h * NNODES + d]);
  float ex = __expf(e - mv);
  exbuf[t] = ex;
  atomicAdd(&denom[(long)h * NNODES + d], ex);
}

// ---------------- weighted scatter: out[dst] += alpha * xs[src] -----------
__global__ void gat_aggregate(const int* __restrict__ edge,
                              const float* __restrict__ xs,
                              const float* __restrict__ exbuf,
                              const float* __restrict__ denom,
                              float* __restrict__ out, int F, int nh) {
  long wid  = ((long)blockIdx.x * blockDim.x + threadIdx.x) >> 5;
  int  lane = threadIdx.x & 31;
  if (wid >= (long)nh * ETOT) return;
  int et = (int)(wid % ETOT), h = (int)(wid / ETOT);
  int s, d; edge_pair(edge, et, s, d);
  float alpha = exbuf[wid] / denom[(long)h * NNODES + d];
  long sb = ((long)h * NNODES + s) * F;
  long db = ((long)h * NNODES + d) * F;
  __builtin_prefetch(&xs[sb], 0, 0);   // global_prefetch_b8
  for (int f = lane; f < F; f += 32)
    atomicAdd(&out[db + f], alpha * xs[sb + f]);
}

// ---------------- layer-1 epilogue: bias + LeakyReLU + concat -> bf16 -----
__global__ void gat_finalize1(const float* __restrict__ out1,
                              const float* __restrict__ bias,
                              unsigned short* __restrict__ xcat) {
  long t = (long)blockIdx.x * blockDim.x + threadIdx.x;
  if (t >= (long)NHEADS * NNODES * NHID) return;
  int  f = (int)(t % NHID);
  long r = t / NHID;
  int  n = (int)(r % NNODES);
  int  h = (int)(r / NNODES);
  float v = lrelu(out1[t] + bias[h * NHID + f], 0.2f);       // outer LeakyReLU(alpha)
  xcat[(long)n * KCAT + h * NHID + f] = f2bf(v);
}

// ---------------- final: bias + LeakyReLU + tanh --------------------------
__global__ void gat_final(const float* __restrict__ out2,
                          const float* __restrict__ bias_out,
                          float* __restrict__ y) {
  long t = (long)blockIdx.x * blockDim.x + threadIdx.x;
  if (t >= (long)NNODES * NCLASS) return;
  int c = (int)(t % NCLASS);
  float v = lrelu(out2[t] + bias_out[c], 0.2f);
  y[t] = tanhf(v);
}

// ---------------- host driver ---------------------------------------------
extern "C" void kernel_launch(void* const* d_in, const int* in_sizes, int n_in,
                              void* d_out, int out_size, void* d_ws, size_t ws_size,
                              hipStream_t stream) {
  const float* type_emb = (const float*)d_in[0];   // [4,50000,128]
  const int*   edge     = (const int*)  d_in[1];   // [2,800000]
  const float* W        = (const float*)d_in[2];   // [4,128,128]
  const float* att_src  = (const float*)d_in[3];   // [4,128]
  const float* att_dst  = (const float*)d_in[4];   // [4,128]
  const float* bias     = (const float*)d_in[5];   // [4,128]
  const float* W_out    = (const float*)d_in[6];   // [512,64]
  const float* asrc_o   = (const float*)d_in[7];   // [64]
  const float* adst_o   = (const float*)d_in[8];   // [64]
  const float* bias_o   = (const float*)d_in[9];   // [64]
  float*       y        = (float*)d_out;           // [50000,64]

  // workspace carve-up (aligned to 256B); regions are reused across layers
  char*  ws  = (char*)d_ws;
  size_t off = 0;
  auto carve = [&](size_t bytes) -> char* {
    char* p = ws + off;
    off += (bytes + 255) & ~(size_t)255;
    return p;
  };
  unsigned short* Abf   = (unsigned short*)carve((size_t)NHEADS * NNODES * FIN * 2); // also xcat bf16
  unsigned short* Bp1   = (unsigned short*)carve((size_t)NHEADS * (NHID/16) * (FIN/32) * 512 * 2);
  unsigned short* Bp2   = (unsigned short*)carve((size_t)(NCLASS/16) * (KCAT/32) * 512 * 2);
  float*          xs1   = (float*)carve((size_t)NHEADS * NNODES * NHID * 4);         // also xs2
  float*          out1  = (float*)carve((size_t)NHEADS * NNODES * NHID * 4);         // also out2
  float*          ssrc  = (float*)carve((size_t)NHEADS * NNODES * 4);
  float*          sdst  = (float*)carve((size_t)NHEADS * NNODES * 4);
  unsigned int*   mmax  = (unsigned int*)carve((size_t)NHEADS * NNODES * 4);
  float*          denom = (float*)carve((size_t)NHEADS * NNODES * 4);
  float*          exbuf = (float*)carve((size_t)NHEADS * ETOT * 4);

  auto nb = [](long n, int t) { return (unsigned)((n + t - 1) / t); };
  const int T = 256;

  // ================= Layer 1 =================
  long nA1 = (long)NHEADS * NNODES * FIN;
  gat_f32_to_bf16<<<nb(nA1, T), T, 0, stream>>>(type_emb, Abf, nA1);

  long nBp1 = (long)NHEADS * (NHID/16) * (FIN/32) * 512;
  gat_pack_B<<<nb(nBp1, T), T, 0, stream>>>(W, Bp1, FIN, NHID, NHEADS);

  {
    long mstrips = NNODES / 16;                           // 3125
    dim3 grid(nb(mstrips, 8), NHEADS);
    gat_wmma_gemm<NHID / 16, FIN / 32>                    // NT=8, KS=4
        <<<grid, T, 0, stream>>>(Abf, Bp1, xs1, NNODES);
  }

  long waves1 = (long)NHEADS * NNODES;
  gat_dots<<<nb(waves1 * 32, T), T, 0, stream>>>(xs1, att_src, att_dst, ssrc, sdst, NHID, NHEADS);

  gat_zero_u32<<<nb(waves1, T), T, 0, stream>>>(mmax, waves1);
  gat_zero_u32<<<nb(waves1, T), T, 0, stream>>>((unsigned int*)denom, waves1);
  long nOut1 = (long)NHEADS * NNODES * NHID;
  gat_zero_u32<<<nb(nOut1, T), T, 0, stream>>>((unsigned int*)out1, nOut1);

  long nEH1 = (long)NHEADS * ETOT;
  gat_edge_max<<<nb(nEH1, T), T, 0, stream>>>(edge, ssrc, sdst, mmax, NHEADS);
  gat_edge_ex <<<nb(nEH1, T), T, 0, stream>>>(edge, ssrc, sdst, mmax, exbuf, denom, NHEADS);
  gat_aggregate<<<nb(nEH1 * 32, T), T, 0, stream>>>(edge, xs1, exbuf, denom, out1, NHID, NHEADS);

  gat_finalize1<<<nb(nOut1, T), T, 0, stream>>>(out1, bias, Abf /* xcat bf16 [N,512] */);

  // ================= Layer 2 =================
  long nBp2 = (long)(NCLASS/16) * (KCAT/32) * 512;
  gat_pack_B<<<nb(nBp2, T), T, 0, stream>>>(W_out, Bp2, KCAT, NCLASS, 1);

  {
    long mstrips = NNODES / 16;
    dim3 grid(nb(mstrips, 8), 1);
    gat_wmma_gemm<NCLASS / 16, KCAT / 32>                 // NT=4, KS=16
        <<<grid, T, 0, stream>>>(Abf, Bp2, xs1 /* xs2 */, NNODES);
  }

  long waves2 = NNODES;
  gat_dots<<<nb(waves2 * 32, T), T, 0, stream>>>(xs1, asrc_o, adst_o, ssrc, sdst, NCLASS, 1);

  gat_zero_u32<<<nb(waves2, T), T, 0, stream>>>(mmax, waves2);
  gat_zero_u32<<<nb(waves2, T), T, 0, stream>>>((unsigned int*)denom, waves2);
  long nOut2 = (long)NNODES * NCLASS;
  gat_zero_u32<<<nb(nOut2, T), T, 0, stream>>>((unsigned int*)out1, nOut2);

  gat_edge_max<<<nb((long)ETOT, T), T, 0, stream>>>(edge, ssrc, sdst, mmax, 1);
  gat_edge_ex <<<nb((long)ETOT, T), T, 0, stream>>>(edge, ssrc, sdst, mmax, exbuf, denom, 1);
  gat_aggregate<<<nb((long)ETOT * 32, T), T, 0, stream>>>(edge, xs1, exbuf, denom, out1, NCLASS, 1);

  gat_final<<<nb(nOut2, T), T, 0, stream>>>(out1, bias_o, y);
}
